// LocalMae_41145786695916
// MI455X (gfx1250) — matrix operand order repaired and checked by reading, not compiled, not run on hardware
//
#include <hip/hip_runtime.h>

// ---------------------------------------------------------------------------
// Local MAE on 56x56 windows, MI455X (gfx1250, wave32, WMMA bf16 16x16x32).
// One window per workgroup (512 WGs x 256 threads = 8 waves).
// Activations LDS-resident (~312KB of the 320KB WGP LDS); weights converted
// once per call to bf16 and transposed (WT[N][K]) -> every B-fragment is two
// contiguous 16B global loads; A-fragments are two contiguous 16B LDS loads.
// Each wave computes an n-tile PAIR sharing one A fragment per k-step
// (2 WMMAs per A load), with compile-time K so k-loops fully unroll.
// V is produced transposed by the QKV GEMM so softmax(S)@V reads contiguous
// runs. Weights ~22MB bf16 -> L2-resident across all 512 blocks.
// ---------------------------------------------------------------------------

typedef __attribute__((ext_vector_type(16))) __bf16 v16bf;
typedef __attribute__((ext_vector_type(8)))  __bf16 v8bf;
typedef __attribute__((ext_vector_type(8)))  float  v8f;

#define WAVES 8
#define NWIN  512           // 8 batches * 8x8 windows
#define DENC  384
#define DDEC  192
#define PDIM  192           // 8*8*3 patch dim

// LDS partition (bytes)
#define XF_OFF   0                         // 64*384 f32  =  98304
#define HB_OFF   (XF_OFF + 64*384*4)       // 64*384 bf16 =  49152
#define QKV_OFF  (HB_OFF + 64*384*2)       // 64*1152 bf16= 147456 (QK row-major + VT)
#define SF_OFF   (QKV_OFF + 64*1152*2)     // 64*64 f32   =  16384
#define SB_OFF   (SF_OFF + 64*64*4)        // 64*64 bf16  =   8192
#define RED_OFF  (SB_OFF + 64*64*2)        // 64 f32      =    256
#define SMEM_TOTAL (RED_OFF + 64*4)        // 319744 B <= 320KB

__device__ __forceinline__ __bf16 f2bf(float f) { return (__bf16)f; }

__device__ __forceinline__ float gelu_tanh(float x) {
  float x3 = x * x * x;
  return 0.5f * x * (1.f + tanhf(0.79788456080286535588f * (x + 0.044715f * x3)));
}

__device__ __forceinline__ v8f wmma_bf16(v16bf a, v16bf b, v8f c) {
  return __builtin_amdgcn_wmma_f32_16x16x32_bf16(false, a, false, b, (short)0, c, false, false);
}

// ---- WMMA fragment loaders (CDNA5 ISA 16-bit layouts) -----------
// A (16x32): lane<16 -> M=lane, K={0..7,16..23}; lane>=16 -> K={8..15,24..31}.
__device__ __forceinline__ v16bf load_a(const __bf16* A, int lda, int m0, int k0, int lane) {
  const __bf16* p = A + (size_t)(m0 + (lane & 15)) * lda + k0 + ((lane >> 4) << 3);
  union { v16bf v; struct { v8bf lo, hi; } s; } u;
  u.s.lo = *(const v8bf*)p;          // K = h8 .. h8+7
  u.s.hi = *(const v8bf*)(p + 16);   // K = 16+h8 .. 16+h8+7
  return u.v;
}
// B (32x16) from an [n][k] matrix (transposed weights WT, or token/dh-major
// K^T / V^T): K={h8..h8+7, 16+h8..16+h8+7} per lane = two 16B loads.
__device__ __forceinline__ v16bf load_b_rows(const __bf16* M, int ldm, int k0, int n0, int lane) {
  const __bf16* p = M + (size_t)(n0 + (lane & 15)) * ldm + k0 + ((lane >> 4) << 3);
  union { v16bf v; struct { v8bf lo, hi; } s; } u;
  u.s.lo = *(const v8bf*)p;
  u.s.hi = *(const v8bf*)(p + 16);
  return u.v;
}

// ---- GEMMs: OUT[64 x N] = A[64 x K] * W[K x N] + bias ----------------------
// Weights transposed: WT[N][K], row stride ldt. Each wave handles an n-tile
// pair sharing the A fragment (2 WMMAs per A load). N/16 must be even.
template<int K>
__device__ __forceinline__ void gemm_f32(const __bf16* A, int lda,
                                         const __bf16* WT, int ldt, const float* bias,
                                         float* OUT, int ldo, int N, bool accum) {
  int lane = threadIdx.x & 31;
  int pairs = (N >> 5) << 2;
  for (int tp = (threadIdx.x >> 5); tp < pairs; tp += WAVES) {
    int mt = (tp & 3) << 4, nt0 = (tp >> 2) << 5, nt1 = nt0 + 16;
    int nn = lane & 15, mb = mt + ((lane >> 4) << 3);
    v8f acc0 = {}, acc1 = {};
    if (accum) {
#pragma unroll
      for (int r = 0; r < 8; ++r) {
        acc0[r] = OUT[(mb + r) * ldo + nt0 + nn];
        acc1[r] = OUT[(mb + r) * ldo + nt1 + nn];
      }
    }
    if (lane == 0) {
      __builtin_prefetch(WT + (size_t)nt0 * ldt, 0, 3);
      __builtin_prefetch(WT + (size_t)nt1 * ldt, 0, 3);
    }
#pragma unroll
    for (int k0 = 0; k0 < K; k0 += 32) {
      v16bf a  = load_a(A, lda, mt, k0, lane);
      v16bf b0 = load_b_rows(WT, ldt, k0, nt0, lane);
      v16bf b1 = load_b_rows(WT, ldt, k0, nt1, lane);
      acc0 = wmma_bf16(a, b0, acc0);
      acc1 = wmma_bf16(a, b1, acc1);
    }
    float bv0 = bias ? bias[nt0 + nn] : 0.f;
    float bv1 = bias ? bias[nt1 + nn] : 0.f;
#pragma unroll
    for (int r = 0; r < 8; ++r) {
      OUT[(mb + r) * ldo + nt0 + nn] = acc0[r] + bv0;
      OUT[(mb + r) * ldo + nt1 + nn] = acc1[r] + bv1;
    }
  }
}

template<int K>
__device__ __forceinline__ void gemm_bf16(const __bf16* A, int lda,
                                          const __bf16* WT, int ldt, const float* bias,
                                          __bf16* OUT, int ldo, int N, bool act) {
  int lane = threadIdx.x & 31;
  int pairs = (N >> 5) << 2;
  for (int tp = (threadIdx.x >> 5); tp < pairs; tp += WAVES) {
    int mt = (tp & 3) << 4, nt0 = (tp >> 2) << 5, nt1 = nt0 + 16;
    int nn = lane & 15, mb = mt + ((lane >> 4) << 3);
    v8f acc0 = {}, acc1 = {};
    if (lane == 0) {
      __builtin_prefetch(WT + (size_t)nt0 * ldt, 0, 3);
      __builtin_prefetch(WT + (size_t)nt1 * ldt, 0, 3);
    }
#pragma unroll
    for (int k0 = 0; k0 < K; k0 += 32) {
      v16bf a  = load_a(A, lda, mt, k0, lane);
      v16bf b0 = load_b_rows(WT, ldt, k0, nt0, lane);
      v16bf b1 = load_b_rows(WT, ldt, k0, nt1, lane);
      acc0 = wmma_bf16(a, b0, acc0);
      acc1 = wmma_bf16(a, b1, acc1);
    }
    float bv0 = bias[nt0 + nn], bv1 = bias[nt1 + nn];
#pragma unroll
    for (int r = 0; r < 8; ++r) {
      float v0 = acc0[r] + bv0, v1 = acc1[r] + bv1;
      OUT[(mb + r) * ldo + nt0 + nn] = f2bf(act ? gelu_tanh(v0) : v0);
      OUT[(mb + r) * ldo + nt1 + nn] = f2bf(act ? gelu_tanh(v1) : v1);
    }
  }
}

// QKV GEMM: Q,K row-major into QK[64 x 2Dm]; V transposed into VT[Dm x 64]
// (per-lane 8 C-rows = 8 consecutive tokens -> one 16B packed LDS store).
template<int Dm>
__device__ __forceinline__ void gemm_qkv(const __bf16* A, const __bf16* WT,
                                         const float* bias, __bf16* QK, __bf16* VT) {
  constexpr int D3 = 3 * Dm, D2 = 2 * Dm;
  int lane = threadIdx.x & 31;
  int pairs = (D3 >> 5) << 2;
  for (int tp = (threadIdx.x >> 5); tp < pairs; tp += WAVES) {
    int mt = (tp & 3) << 4, nt0 = (tp >> 2) << 5, nt1 = nt0 + 16;
    int nn = lane & 15, mb = mt + ((lane >> 4) << 3);
    v8f acc0 = {}, acc1 = {};
    if (lane == 0) {
      __builtin_prefetch(WT + (size_t)nt0 * Dm, 0, 3);
      __builtin_prefetch(WT + (size_t)nt1 * Dm, 0, 3);
    }
#pragma unroll
    for (int k0 = 0; k0 < Dm; k0 += 32) {
      v16bf a  = load_a(A, Dm, mt, k0, lane);
      v16bf b0 = load_b_rows(WT, Dm, k0, nt0, lane);
      v16bf b1 = load_b_rows(WT, Dm, k0, nt1, lane);
      acc0 = wmma_bf16(a, b0, acc0);
      acc1 = wmma_bf16(a, b1, acc1);
    }
#pragma unroll
    for (int half = 0; half < 2; ++half) {
      int ntX = half ? nt1 : nt0;
      v8f accX = half ? acc1 : acc0;
      float bv = bias[ntX + nn];
      if (ntX < D2) {                  // Q or K tile: row-major
#pragma unroll
        for (int r = 0; r < 8; ++r) QK[(mb + r) * D2 + ntX + nn] = f2bf(accX[r] + bv);
      } else {                         // V tile: transposed packed store
        v8bf pk;
#pragma unroll
        for (int r = 0; r < 8; ++r) pk[r] = f2bf(accX[r] + bv);
        *(v8bf*)(VT + (size_t)(ntX - D2 + nn) * 64 + mb) = pk;
      }
    }
  }
}

// ---- LayerNorm: 64 rows x Dm, 4 threads/row with intra-quad shuffles -------
template<int Dm>
__device__ __forceinline__ void layernorm64(const float* X, const float* g,
                                            const float* b, __bf16* H) {
  int r = threadIdx.x >> 2, q = threadIdx.x & 3;
  const float* row = X + (size_t)r * Dm;
  float mu = 0.f;
  for (int j = q; j < Dm; j += 4) mu += row[j];
  mu += __shfl_xor(mu, 1, 4);
  mu += __shfl_xor(mu, 2, 4);
  mu *= (1.f / Dm);
  float var = 0.f;
  for (int j = q; j < Dm; j += 4) { float d = row[j] - mu; var += d * d; }
  var += __shfl_xor(var, 1, 4);
  var += __shfl_xor(var, 2, 4);
  var *= (1.f / Dm);
  float inv = rsqrtf(var + 1e-6f);
  __bf16* h = H + (size_t)r * Dm;
  for (int j = q; j < Dm; j += 4) h[j] = f2bf((row[j] - mu) * inv * g[j] + b[j]);
}

// ---- Multi-head attention ---------------------------------------------------
// QK: [64 x 2Dm] row-major (Q | K), VT: [Dm x 64] transposed V.
template<int Dm, int nh>
__device__ __forceinline__ void attention(const __bf16* QK, const __bf16* VT, __bf16* O,
                                          float* SF, __bf16* SB) {
  constexpr int D2 = 2 * Dm, dh = Dm / nh;
  int lane = threadIdx.x & 31;
  float scale = rsqrtf((float)dh);
  for (int h = 0; h < nh; ++h) {
    const __bf16* Q  = QK + h * dh;
    const __bf16* Km = QK + Dm + h * dh;          // [token][dh] == [n][k] for B
    const __bf16* Vh = VT + (size_t)h * dh * 64;  // [dh][token] == [n][k] for B
    // S[64x64] = Q * K^T  (8 tile-pairs: exactly one per wave)
    for (int tp = (threadIdx.x >> 5); tp < 8; tp += WAVES) {
      int mt = (tp & 3) << 4, nt0 = (tp >> 2) << 5, nt1 = nt0 + 16;
      int nn = lane & 15, mb = mt + ((lane >> 4) << 3);
      v8f acc0 = {}, acc1 = {};
#pragma unroll
      for (int k0 = 0; k0 < dh; k0 += 32) {
        v16bf a  = load_a(Q, D2, mt, k0, lane);
        v16bf b0 = load_b_rows(Km, D2, k0, nt0, lane);
        v16bf b1 = load_b_rows(Km, D2, k0, nt1, lane);
        acc0 = wmma_bf16(a, b0, acc0);
        acc1 = wmma_bf16(a, b1, acc1);
      }
#pragma unroll
      for (int r = 0; r < 8; ++r) {
        SF[(mb + r) * 64 + nt0 + nn] = acc0[r];
        SF[(mb + r) * 64 + nt1 + nn] = acc1[r];
      }
    }
    __syncthreads();
    // softmax over the 49 valid keys; zero padded columns
    if (threadIdx.x < 64) {
      float* row = SF + threadIdx.x * 64;
      float mx = -3.4e38f;
      for (int j = 0; j < 49; ++j) mx = fmaxf(mx, row[j] * scale);
      float s = 0.f;
      for (int j = 0; j < 49; ++j) { float e = __expf(row[j] * scale - mx); row[j] = e; s += e; }
      float inv = 1.f / s;
      __bf16* rb = SB + threadIdx.x * 64;
      for (int j = 0; j < 64; ++j) rb[j] = f2bf(j < 49 ? row[j] * inv : 0.f);
    }
    __syncthreads();
    // O[:, h*dh:] = softmax(S) @ V    (K = 64 padded tokens, B from VT)
    constexpr int opairs = (dh >> 5) << 2;
    for (int tp = (threadIdx.x >> 5); tp < opairs; tp += WAVES) {
      int mt = (tp & 3) << 4, nt0 = (tp >> 2) << 5, nt1 = nt0 + 16;
      int nn = lane & 15, mb = mt + ((lane >> 4) << 3);
      v8f acc0 = {}, acc1 = {};
#pragma unroll
      for (int k0 = 0; k0 < 64; k0 += 32) {
        v16bf a  = load_a(SB, 64, mt, k0, lane);
        v16bf b0 = load_b_rows(Vh, 64, k0, nt0, lane);
        v16bf b1 = load_b_rows(Vh, 64, k0, nt1, lane);
        acc0 = wmma_bf16(a, b0, acc0);
        acc1 = wmma_bf16(a, b1, acc1);
      }
#pragma unroll
      for (int r = 0; r < 8; ++r) {
        O[(mb + r) * Dm + h * dh + nt0 + nn] = f2bf(acc0[r]);
        O[(mb + r) * Dm + h * dh + nt1 + nn] = f2bf(acc1[r]);
      }
    }
    __syncthreads();
  }
}

// ---- transformer block ------------------------------------------------------
struct BlockW {
  const __bf16 *qkv_w, *proj_w, *fc1_w, *fc2_w;   // transposed bf16
  const float  *qkv_b, *proj_b, *fc1_b, *fc2_b, *ln1_g, *ln1_b, *ln2_g, *ln2_b;
};

template<int Dm, int nh>
__device__ __forceinline__ void xblock(float* XF, __bf16* HB, __bf16* QKVB,
                                       float* SF, __bf16* SB,
                                       const BlockW& bw, int layer) {
  constexpr int D3 = 3 * Dm, D4 = 4 * Dm;
  constexpr int CH = (D4 > 1152) ? (D4 / 2) : D4;   // enc: 2x768, dec: 1x768
  const __bf16* qw  = bw.qkv_w  + (size_t)layer * Dm * D3;
  const __bf16* pw  = bw.proj_w + (size_t)layer * Dm * Dm;
  const __bf16* f1w = bw.fc1_w  + (size_t)layer * Dm * D4;
  const __bf16* f2w = bw.fc2_w  + (size_t)layer * D4 * Dm;
  const float* qb  = bw.qkv_b  + layer * D3;
  const float* pb  = bw.proj_b + layer * Dm;
  const float* f1b = bw.fc1_b  + layer * D4;
  const float* f2b = bw.fc2_b  + layer * Dm;
  __bf16* VT = QKVB + (size_t)64 * 2 * Dm;

  layernorm64<Dm>(XF, bw.ln1_g + layer * Dm, bw.ln1_b + layer * Dm, HB);
  __syncthreads();
  gemm_qkv<Dm>(HB, qw, qb, QKVB, VT);
  __syncthreads();
  attention<Dm, nh>(QKVB, VT, HB, SF, SB);
  gemm_f32<Dm>(HB, Dm, pw, Dm, pb, XF, Dm, Dm, true);   // x += o @ proj + b
  __syncthreads();
  layernorm64<Dm>(XF, bw.ln2_g + layer * Dm, bw.ln2_b + layer * Dm, HB);
  __syncthreads();
  for (int c0 = 0; c0 < D4; c0 += CH) {
    // fc1 columns c0..c0+CH: WT rows c0.. -> offset c0*Dm
    gemm_bf16<Dm>(HB, Dm, f1w + (size_t)c0 * Dm, Dm, f1b + c0, QKVB, CH, CH, true);
    __syncthreads();
    // fc2 k-range c0..c0+CH: WT column offset c0 within each row (stride D4)
    gemm_f32<CH>(QKVB, CH, f2w + c0, D4, (c0 == 0) ? f2b : nullptr, XF, Dm, Dm, true);
    __syncthreads();
  }
}

// ---- kernel args ------------------------------------------------------------
struct MaeArgs {
  const __bf16* patches;     // [512][64][192] bf16 (padded rows zero)
  const float*  pm;          // [512][64]
  const __bf16* embed_w; const float *embed_b, *pos, *mask_token;
  BlockW enc, dec;
  const float *enc_ng, *enc_nb;
  const __bf16* dec_w; const float *dec_b, *dec_pos;
  const float *dec_ng, *dec_nb;
  const __bf16* head_w; const float* head_b;
  const float* target;
  float* pred;               // [512][49][192]
  float* loss_w;             // [512]
};

// ---- kernels ----------------------------------------------------------------
// Convert f32 -> bf16 and transpose each K x N layer to WT[N][K].
__global__ void k_cvt_t(const float* __restrict__ src, __bf16* __restrict__ dst,
                        int K, int N, int total) {
  int i = blockIdx.x * 256 + threadIdx.x;
  if (i >= total) return;
  int kn = K * N;
  int l = i / kn, rem = i - l * kn;
  int k = rem / N, n = rem - k * N;
  dst[(size_t)l * kn + (size_t)n * K + k] = f2bf(src[i]);
}

__global__ void k_prep(const float* __restrict__ imgs, const float* __restrict__ mask,
                       float* __restrict__ pm, float* __restrict__ sel,
                       __bf16* __restrict__ patches) {
  __shared__ int cnt;
  if (threadIdx.x == 0) cnt = 0;
  __syncthreads();
  int w = blockIdx.x, b = w >> 6, wy = (w >> 3) & 7, wx = w & 7;
  float p = 0.f;
  if (threadIdx.x < 49) {
    int ty = threadIdx.x / 7, tx = threadIdx.x - ty * 7;
    const float* m0 = mask + (size_t)b * 448 * 448;
    int y0 = wy * 56 + ty * 8, x0 = wx * 56 + tx * 8;
    int c = 0;
    for (int py = 0; py < 8; ++py)
      for (int px = 0; px < 8; ++px) {
        float m = m0[(y0 + py) * 448 + x0 + px];
        c += (m < 0.15f || m > 0.85f) ? 1 : 0;
      }
    p = (c >= 16) ? 1.f : 0.f;                // PATCH_THR / P^2
    if (p != 0.f) atomicAdd(&cnt, 1);
  }
  if (threadIdx.x < 64) pm[(size_t)w * 64 + threadIdx.x] = (threadIdx.x < 49) ? p : 0.f;
  __syncthreads();
  if (threadIdx.x == 0) sel[w] = (cnt >= 8) ? 1.f : 0.f;   // 64*cnt > 448
  __bf16* pw = patches + (size_t)w * 64 * 192;
  for (int i = threadIdx.x; i < 64 * 192; i += 256) {
    int t = i / 192, j = i - t * 192;
    float v = 0.f;
    if (t < 49) {
      int ty = t / 7, tx = t - ty * 7;
      int c = j % 3, px = (j / 3) & 7, py = j / 24;
      v = imgs[(((size_t)(b * 3 + c)) * 448 + (wy * 56 + ty * 8 + py)) * 448 +
               (wx * 56 + tx * 8 + px)];
    }
    pw[i] = f2bf(v);
  }
}

__global__ void __launch_bounds__(256) k_mae(MaeArgs A) {
  extern __shared__ char smem[];
  float*  XF   = (float*)(smem + XF_OFF);
  __bf16* HB   = (__bf16*)(smem + HB_OFF);
  __bf16* QKVB = (__bf16*)(smem + QKV_OFF);
  float*  SF   = (float*)(smem + SF_OFF);
  __bf16* SB   = (__bf16*)(smem + SB_OFF);
  float*  red  = (float*)(smem + RED_OFF);

  int w = blockIdx.x;
  const float* pmw = A.pm + (size_t)w * 64;

  // stage patches into LDS (16B vector copies), embed GEMM
  const uint4* Pv = (const uint4*)(A.patches + (size_t)w * 64 * 192);
  uint4* HBv = (uint4*)HB;
  for (int i = threadIdx.x; i < 64 * 192 * 2 / 16; i += 256) HBv[i] = Pv[i];
  __syncthreads();
  gemm_f32<192>(HB, 192, A.embed_w, 192, A.embed_b, XF, DENC, DENC, false);
  __syncthreads();
  // + pos, mask-token mix, zero padded rows
  for (int i = threadIdx.x; i < 64 * DENC; i += 256) {
    int t = i / DENC, d = i - t * DENC;
    float v = 0.f;
    if (t < 49) {
      float p = pmw[t];
      v = (XF[i] + A.pos[t * DENC + d]) * (1.f - p) + A.mask_token[d] * p;
    }
    XF[i] = v;
  }
  __syncthreads();

  for (int l = 0; l < 6; ++l) xblock<DENC, 6>(XF, HB, QKVB, SF, SB, A.enc, l);

  layernorm64<DENC>(XF, A.enc_ng, A.enc_nb, HB);
  __syncthreads();
  // decoder embed: y = x @ dec_w + dec_b (+ dec_pos)
  gemm_f32<DENC>(HB, DENC, A.dec_w, DENC, A.dec_b, XF, DDEC, DDEC, false);
  __syncthreads();
  for (int i = threadIdx.x; i < 64 * DDEC; i += 256) {
    int t = i / DDEC, d = i - t * DDEC;
    XF[i] = (t < 49) ? XF[i] + A.dec_pos[t * DDEC + d] : 0.f;
  }
  __syncthreads();

  for (int l = 0; l < 2; ++l) xblock<DDEC, 6>(XF, HB, QKVB, SF, SB, A.dec, l);

  layernorm64<DDEC>(XF, A.dec_ng, A.dec_nb, HB);
  __syncthreads();
  // head: pred = y @ head_w + head_b (into reused QKV LDS as f32)
  float* PRED = (float*)QKVB;                // 64*192 f32 fits in 144KB region
  gemm_f32<DDEC>(HB, DDEC, A.head_w, DDEC, A.head_b, PRED, PDIM, PDIM, false);
  __syncthreads();

  // emit pred patches (16B vector copies) + per-window masked-MSE loss
  uint4* predv = (uint4*)(A.pred + (size_t)w * 49 * PDIM);
  const uint4* PREDv = (const uint4*)PRED;
  for (int i = threadIdx.x; i < 49 * PDIM * 4 / 16; i += 256) predv[i] = PREDv[i];
  int b = w >> 6, wy = (w >> 3) & 7, wx = w & 7;
  if (threadIdx.x < 64) red[threadIdx.x] = 0.f;
  __syncthreads();
  if (threadIdx.x < 49) {
    int t = threadIdx.x, ty = t / 7, tx = t - ty * 7;
    float ss = 0.f;
    for (int j = 0; j < PDIM; ++j) {
      int c = j % 3, px = (j / 3) & 7, py = j / 24;
      float tg = A.target[(((size_t)(b * 3 + c)) * 448 + (wy * 56 + ty * 8 + py)) * 448 +
                          (wx * 56 + tx * 8 + px)];
      float d = PRED[t * PDIM + j] - tg;
      ss += d * d;
    }
    red[t] = ss * (1.f / PDIM);
  }
  __syncthreads();
  if (threadIdx.x == 0) {
    float sl = 0.f, sp = 0.f;
    for (int t = 0; t < 49; ++t) { sl += red[t] * pmw[t]; sp += pmw[t]; }
    A.loss_w[w] = sl / fmaxf(sp, 1.f);
  }
}

__global__ void k_out(const float* __restrict__ imgs, const float* __restrict__ pm,
                      const float* __restrict__ sel, const float* __restrict__ pred,
                      float* __restrict__ out, int total) {
  int idx = blockIdx.x * 256 + threadIdx.x;
  if (idx >= total) return;
  int x = idx % 448, y = (idx / 448) % 448;
  int c = (idx / (448 * 448)) % 3, b = idx / (3 * 448 * 448);
  int wy = y / 56, wx = x / 56;
  int w = (b << 6) | (wy << 3) | wx;
  int yy = y - wy * 56, xx = x - wx * 56;
  int ty = yy >> 3, tx = xx >> 3, t = ty * 7 + tx;
  float rec = pm[(size_t)w * 64 + t];
  float img = imgs[idx];
  int py = yy & 7, px = xx & 7;
  float pr = pred[((size_t)w * 49 + t) * PDIM + (py * 8 + px) * 3 + c];
  out[idx] = (sel[w] > 0.f) ? img * (1.f - rec) + pr * rec : img;
}

__global__ void k_loss(const float* __restrict__ loss_w, const float* __restrict__ sel,
                       float* __restrict__ out_loss) {
  __shared__ float sL[256], sS[256];
  float l = 0.f, s = 0.f;
  for (int w = threadIdx.x; w < NWIN; w += 256) { l += loss_w[w] * sel[w]; s += sel[w]; }
  sL[threadIdx.x] = l; sS[threadIdx.x] = s;
  __syncthreads();
  for (int st = 128; st > 0; st >>= 1) {
    if (threadIdx.x < st) { sL[threadIdx.x] += sL[threadIdx.x + st];
                            sS[threadIdx.x] += sS[threadIdx.x + st]; }
    __syncthreads();
  }
  if (threadIdx.x == 0) out_loss[0] = sL[0] / fmaxf(sS[0], 1.f);
}

// ---------------------------------------------------------------------------
extern "C" void kernel_launch(void* const* d_in, const int* in_sizes, int n_in,
                              void* d_out, int out_size, void* d_ws, size_t ws_size,
                              hipStream_t stream) {
  (void)in_sizes; (void)n_in; (void)ws_size;
  // d_in: imgs, mask, target, then params flattened in sorted-key (jax tree) order
  const float* imgs   = (const float*)d_in[0];
  const float* mask   = (const float*)d_in[1];
  const float* target = (const float*)d_in[2];
  const float* dec_fc1_b = (const float*)d_in[3];
  const float* dec_fc1_w = (const float*)d_in[4];
  const float* dec_fc2_b = (const float*)d_in[5];
  const float* dec_fc2_w = (const float*)d_in[6];
  const float* dec_ln1_b = (const float*)d_in[7];
  const float* dec_ln1_g = (const float*)d_in[8];
  const float* dec_ln2_b = (const float*)d_in[9];
  const float* dec_ln2_g = (const float*)d_in[10];
  const float* dec_proj_b = (const float*)d_in[11];
  const float* dec_proj_w = (const float*)d_in[12];
  const float* dec_qkv_b  = (const float*)d_in[13];
  const float* dec_qkv_w  = (const float*)d_in[14];
  const float* dec_b   = (const float*)d_in[15];
  const float* dec_nb  = (const float*)d_in[16];
  const float* dec_ng  = (const float*)d_in[17];
  const float* dec_pos = (const float*)d_in[18];
  const float* dec_w   = (const float*)d_in[19];
  const float* embed_b = (const float*)d_in[20];
  const float* embed_w = (const float*)d_in[21];
  const float* enc_fc1_b = (const float*)d_in[22];
  const float* enc_fc1_w = (const float*)d_in[23];
  const float* enc_fc2_b = (const float*)d_in[24];
  const float* enc_fc2_w = (const float*)d_in[25];
  const float* enc_ln1_b = (const float*)d_in[26];
  const float* enc_ln1_g = (const float*)d_in[27];
  const float* enc_ln2_b = (const float*)d_in[28];
  const float* enc_ln2_g = (const float*)d_in[29];
  const float* enc_proj_b = (const float*)d_in[30];
  const float* enc_proj_w = (const float*)d_in[31];
  const float* enc_qkv_b  = (const float*)d_in[32];
  const float* enc_qkv_w  = (const float*)d_in[33];
  const float* enc_nb = (const float*)d_in[34];
  const float* enc_ng = (const float*)d_in[35];
  const float* head_b = (const float*)d_in[36];
  const float* head_w = (const float*)d_in[37];
  const float* mask_token = (const float*)d_in[38];
  const float* pos = (const float*)d_in[39];

  // workspace layout
  char* ws = (char*)d_ws;
  size_t off = 0;
  auto take = [&](size_t bytes) {
    char* p = ws + off;
    off += (bytes + 255) & ~(size_t)255;
    return p;
  };
  // convert + transpose each K x N weight (per layer) to bf16 WT[N][K]
  auto cvt_t = [&](const float* src, int K, int N, int layers) {
    size_t n = (size_t)layers * K * N;
    __bf16* dst = (__bf16*)take(n * 2);
    k_cvt_t<<<dim3((unsigned)((n + 255) / 256)), dim3(256), 0, stream>>>(src, dst, K, N, (int)n);
    return dst;
  };

  __bf16* embed_w_bf = cvt_t(embed_w, 192, 384, 1);
  __bf16* eqkv = cvt_t(enc_qkv_w, 384, 1152, 6);
  __bf16* eprj = cvt_t(enc_proj_w, 384, 384, 6);
  __bf16* efc1 = cvt_t(enc_fc1_w, 384, 1536, 6);
  __bf16* efc2 = cvt_t(enc_fc2_w, 1536, 384, 6);
  __bf16* dw   = cvt_t(dec_w, 384, 192, 1);
  __bf16* dqkv = cvt_t(dec_qkv_w, 192, 576, 2);
  __bf16* dprj = cvt_t(dec_proj_w, 192, 192, 2);
  __bf16* dfc1 = cvt_t(dec_fc1_w, 192, 768, 2);
  __bf16* dfc2 = cvt_t(dec_fc2_w, 768, 192, 2);
  __bf16* hw   = cvt_t(head_w, 192, 192, 1);

  __bf16* patches = (__bf16*)take((size_t)NWIN * 64 * 192 * 2);
  float* pm    = (float*)take((size_t)NWIN * 64 * 4);
  float* sel   = (float*)take((size_t)NWIN * 4);
  float* pred  = (float*)take((size_t)NWIN * 49 * 192 * 4);
  float* lossw = (float*)take((size_t)NWIN * 4);

  k_prep<<<dim3(NWIN), dim3(256), 0, stream>>>(imgs, mask, pm, sel, patches);

  MaeArgs A;
  A.patches = patches; A.pm = pm;
  A.embed_w = embed_w_bf; A.embed_b = embed_b; A.pos = pos; A.mask_token = mask_token;
  A.enc = { eqkv, eprj, efc1, efc2,
            enc_qkv_b, enc_proj_b, enc_fc1_b, enc_fc2_b,
            enc_ln1_g, enc_ln1_b, enc_ln2_g, enc_ln2_b };
  A.dec = { dqkv, dprj, dfc1, dfc2,
            dec_qkv_b, dec_proj_b, dec_fc1_b, dec_fc2_b,
            dec_ln1_g, dec_ln1_b, dec_ln2_g, dec_ln2_b };
  A.enc_ng = enc_ng; A.enc_nb = enc_nb;
  A.dec_w = dw; A.dec_b = dec_b; A.dec_pos = dec_pos;
  A.dec_ng = dec_ng; A.dec_nb = dec_nb;
  A.head_w = hw; A.head_b = head_b;
  A.target = target; A.pred = pred; A.loss_w = lossw;

  k_mae<<<dim3(NWIN), dim3(256), SMEM_TOTAL, stream>>>(A);

  const int total = 8 * 3 * 448 * 448;
  float* out = (float*)d_out;
  k_out<<<dim3((total + 255) / 256), dim3(256), 0, stream>>>(imgs, pm, sel, pred, out, total);
  k_loss<<<dim3(1), dim3(256), 0, stream>>>(lossw, sel, out + total);
  (void)out_size;
}
